// SelfAttention_81114752352642
// MI455X (gfx1250) — matrix operand order repaired and checked by reading, not compiled, not run on hardware
//
#include <hip/hip_runtime.h>

typedef _Float16 v16h __attribute__((ext_vector_type(16)));
typedef _Float16 v8h  __attribute__((ext_vector_type(8)));
typedef float    v8f  __attribute__((ext_vector_type(8)));
typedef unsigned int v4u __attribute__((ext_vector_type(4)));
typedef int      v8i __attribute__((ext_vector_type(8)));
typedef int      v4i __attribute__((ext_vector_type(4)));

#define WMMA_F32_F16(a, b, c) \
  __builtin_amdgcn_wmma_f32_16x16x32_f16(false, (a), false, (b), (short)0, (c), false, false)

// ---------------------------------------------------------------------------
// TDM: 2-D tile (f16 elements) global -> LDS, one descriptor, TENSORcnt-tracked.
// tensor_d0/d1 = tensor extent (elements), tile_d0/d1 = tile extent,
// stride0 = elements between consecutive dim-1 lines. Tile lands contiguous
// row-major in LDS at lds_off.
// ---------------------------------------------------------------------------
__device__ __forceinline__ void tdm_load_2d_f16(unsigned lds_off, const void* gaddr,
                                                unsigned tensor_d0, unsigned tensor_d1,
                                                unsigned tile_d0, unsigned tile_d1,
                                                unsigned long long stride0) {
  unsigned long long ga = (unsigned long long)gaddr;
  v4u g0;
  g0[0] = 1u;                                   // count=1, user descriptor
  g0[1] = lds_off;                              // lds_addr
  g0[2] = (unsigned)ga;                         // global_addr[31:0]
  g0[3] = (unsigned)(ga >> 32) | 0x80000000u;   // global_addr[56:32] | type=2
  v8i g1;
  g1[0] = 1 << 16;                                                  // data_size=2B
  g1[1] = (int)((tensor_d0 & 0xFFFFu) << 16);                       // tensor_dim0 lo
  g1[2] = (int)((tensor_d0 >> 16) | ((tensor_d1 & 0xFFFFu) << 16)); // d0 hi | d1 lo
  g1[3] = (int)((tensor_d1 >> 16) | (tile_d0 << 16));               // d1 hi | tile0
  g1[4] = (int)(tile_d1 & 0xFFFFu);                                 // tile1 | tile2=0
  g1[5] = (int)(unsigned)(stride0 & 0xFFFFFFFFu);                   // dim0_stride lo
  g1[6] = (int)(unsigned)((stride0 >> 32) & 0xFFFFu);               // stride hi | d1str=0
  g1[7] = 0;
  v4i z4 = {0, 0, 0, 0};
#if __clang_major__ >= 23
  v8i z8 = {0, 0, 0, 0, 0, 0, 0, 0};
  __builtin_amdgcn_tensor_load_to_lds(g0, g1, z4, z4, z8, 0);
#else
  __builtin_amdgcn_tensor_load_to_lds(g0, g1, z4, z4, 0);
#endif
}

__device__ __forceinline__ unsigned lds_off_of(const void* p) {
  // generic pointers to LDS carry the LDS offset in the low 32 bits
  return (unsigned)(unsigned long long)p;
}

// ---------------------------------------------------------------------------
// fp32 -> fp16 conversion (grid-stride)
// ---------------------------------------------------------------------------
__global__ void cvt_f32_f16(const float* __restrict__ in, _Float16* __restrict__ out, int n) {
  int i = blockIdx.x * blockDim.x + threadIdx.x;
  int stride = gridDim.x * blockDim.x;
  for (; i < n; i += stride) out[i] = (_Float16)in[i];
}

// ---------------------------------------------------------------------------
// fp32 [rows,cols] -> fp16 transposed [cols,rows], 32x32 LDS tiles
// ---------------------------------------------------------------------------
__global__ void cvt_transpose_f16(const float* __restrict__ in, _Float16* __restrict__ out,
                                  int rows, int cols) {
  __shared__ _Float16 tile[32][33];
  const int c0 = blockIdx.x * 32, r0 = blockIdx.y * 32;
  const int tx = threadIdx.x, ty = threadIdx.y;  // (32, 8)
#pragma unroll
  for (int i = 0; i < 32; i += 8)
    tile[ty + i][tx] = (_Float16)in[(size_t)(r0 + ty + i) * cols + c0 + tx];
  __syncthreads();
#pragma unroll
  for (int i = 0; i < 32; i += 8)
    out[(size_t)(c0 + ty + i) * rows + r0 + tx] = tile[tx][ty + i];
}

// ---------------------------------------------------------------------------
// Tiled f16 GEMM, f32 accumulate: C[M,N] = A[M,K] * Bt[N,K]^T
// BM=BN=128, BK=32, 256 threads = 8 waves. TDM double-buffered LDS staging.
// mode 0: scatter into Q/K head layout [bh,t,64] f16 and V transposed [bh,64,t]
// mode 1: f32 row-major [M,N] to Cf
// ---------------------------------------------------------------------------
__global__ __launch_bounds__(256)
void gemm_f16f32(const _Float16* __restrict__ A, const _Float16* __restrict__ Bt,
                 int M, int N, int K, int mode,
                 float* __restrict__ Cf,
                 _Float16* __restrict__ Qh, _Float16* __restrict__ Kh,
                 _Float16* __restrict__ VTh) {
  __shared__ _Float16 sA[2][128 * 32];  // [m][k], pitch 32
  __shared__ _Float16 sB[2][128 * 32];  // [n][k], pitch 32 (rows of Bt)

  const int tid  = threadIdx.x;
  const int wid  = tid >> 5;
  const int lane = tid & 31;
  const int half = lane >> 4;
  const int ln16 = lane & 15;
  const int wm   = wid & 3;   // 4 waves along M (32 rows each)
  const int wn   = wid >> 2;  // 2 waves along N (64 cols each)
  const int bm = blockIdx.y, bn = blockIdx.x;

  const _Float16* Abase = A  + (size_t)(bm * 128) * K;
  const _Float16* Bbase = Bt + (size_t)(bn * 128) * K;
  const int NK = K >> 5;

  v8f acc[2][4] = {};

  if (wid == 0) {
    tdm_load_2d_f16(lds_off_of(&sA[0][0]), Abase, K, 128, 32, 128, K);
    tdm_load_2d_f16(lds_off_of(&sB[0][0]), Bbase, K, 128, 32, 128, K);
  }

  for (int it = 0; it < NK; ++it) {
    const int buf = it & 1;
    if (wid == 0) __builtin_amdgcn_s_wait_tensorcnt(0);
    __syncthreads();  // tile `it` visible; all waves done reading buf^1
    if (wid == 0 && it + 1 < NK) {
      const int k0 = (it + 1) << 5;
      tdm_load_2d_f16(lds_off_of(&sA[buf ^ 1][0]), Abase + k0, K, 128, 32, 128, K);
      tdm_load_2d_f16(lds_off_of(&sB[buf ^ 1][0]), Bbase + k0, K, 128, 32, 128, K);
    }

    v16h af[2], bf[4];
#pragma unroll
    for (int tm = 0; tm < 2; ++tm) {
      const _Float16* base = &sA[buf][(wm * 32 + tm * 16 + ln16) * 32];
      v8h lo = *(const v8h*)&base[half * 16];       // K = half*16 .. +7
      v8h hi = *(const v8h*)&base[8 + half * 16];  // K = half*16+8 .. +7
#pragma unroll
      for (int i = 0; i < 8; ++i) { af[tm][i] = lo[i]; af[tm][i + 8] = hi[i]; }
    }
#pragma unroll
    for (int tn = 0; tn < 4; ++tn)
      bf[tn] = *(const v16h*)&sB[buf][(wn * 64 + tn * 16 + ln16) * 32 + half * 16];
#pragma unroll
    for (int tm = 0; tm < 2; ++tm)
#pragma unroll
      for (int tn = 0; tn < 4; ++tn)
        acc[tm][tn] = WMMA_F32_F16(af[tm], bf[tn], acc[tm][tn]);
  }

  // epilogue: `which` is uniform per 16-wide tile (tiles never straddle 1024)
#pragma unroll
  for (int tm = 0; tm < 2; ++tm)
#pragma unroll
    for (int tn = 0; tn < 4; ++tn) {
      const int m_base = bm * 128 + wm * 32 + tm * 16 + half * 8;
      const int n      = bn * 128 + wn * 64 + tn * 16 + ln16;
      if (mode == 0) {
        const int which = n >> 10, c = n & 1023;
        const int h = c >> 6, d = c & 63;
        if (which < 2) {
          _Float16* dst = which ? Kh : Qh;
#pragma unroll
          for (int r = 0; r < 8; ++r) {
            int m = m_base + r;
            int b = m >> 11, t = m & 2047;
            dst[(((size_t)(b * 16 + h)) * 2048 + t) * 64 + d] = (_Float16)acc[tm][tn][r];
          }
        } else {
#pragma unroll
          for (int r = 0; r < 8; ++r) {
            int m = m_base + r;
            int b = m >> 11, t = m & 2047;
            VTh[(((size_t)(b * 16 + h)) * 64 + d) * 2048 + t] = (_Float16)acc[tm][tn][r];
          }
        }
      } else {
#pragma unroll
        for (int r = 0; r < 8; ++r)
          Cf[(size_t)(m_base + r) * N + n] = acc[tm][tn][r];
      }
    }
}

// ---------------------------------------------------------------------------
// Flash attention: block = (b,h, 128 queries), 8 waves x 16 queries each.
// Q,K in [bh,t,64] f16; V transposed [bh,64,t] f16; Y out [b,t,1024] f16.
// TDM double-buffered K/V tiles (32 keys per step).
// ---------------------------------------------------------------------------
__global__ __launch_bounds__(256)
void flash_attn(const _Float16* __restrict__ Qh, const _Float16* __restrict__ Kh,
                const _Float16* __restrict__ VTh, _Float16* __restrict__ Yh) {
  __shared__ _Float16 sK[2][32 * 64];    // [key][d]
  __shared__ _Float16 sVT[2][64 * 32];   // [d][key]
  __shared__ _Float16 sP[8][16 * 32];    // per-wave P scratch [row][key]

  const int tid  = threadIdx.x;
  const int wid  = tid >> 5;
  const int lane = tid & 31;
  const int half = lane >> 4;
  const int ln16 = lane & 15;

  const int bh = blockIdx.y;  // 0..31
  const int b = bh >> 4, h = bh & 15;
  const int qbase = blockIdx.x * 128;

  const _Float16* Kbase  = Kh  + (size_t)bh * 2048 * 64;
  const _Float16* VTbase = VTh + (size_t)bh * 64 * 2048;

  // Q fragments, pre-scaled by 1/sqrt(64)=0.125 (exact in f16)
  const _Float16 scale = (_Float16)0.125f;
  v16h qf[2];
  {
    const int q = qbase + wid * 16 + ln16;
    const _Float16* qrow = Qh + ((size_t)bh * 2048 + q) * 64;
#pragma unroll
    for (int dg = 0; dg < 2; ++dg) {
      v8h lo = *(const v8h*)&qrow[dg * 32 + half * 16];
      v8h hi = *(const v8h*)&qrow[dg * 32 + 8 + half * 16];
#pragma unroll
      for (int i = 0; i < 8; ++i) {
        qf[dg][i]     = lo[i] * scale;
        qf[dg][i + 8] = hi[i] * scale;
      }
    }
  }

  v8f O[4] = {};
  float mrow[8], lrow[8];
#pragma unroll
  for (int r = 0; r < 8; ++r) { mrow[r] = -3.0e38f; lrow[r] = 0.0f; }

  if (wid == 0) {
    tdm_load_2d_f16(lds_off_of(&sK[0][0]),  Kbase,  64,   2048, 64, 32, 64);
    tdm_load_2d_f16(lds_off_of(&sVT[0][0]), VTbase, 2048, 64,   32, 64, 2048);
  }

  for (int it = 0; it < 64; ++it) {
    const int buf = it & 1;
    if (wid == 0) __builtin_amdgcn_s_wait_tensorcnt(0);
    __syncthreads();  // tiles visible; buf^1 no longer being read
    if (wid == 0 && it + 1 < 64) {
      const int kb = (it + 1) << 5;
      tdm_load_2d_f16(lds_off_of(&sK[buf ^ 1][0]),  Kbase + (size_t)kb * 64, 64, 2048, 64, 32, 64);
      tdm_load_2d_f16(lds_off_of(&sVT[buf ^ 1][0]), VTbase + kb, 2048, 64, 32, 64, 2048);
    }

    // S = Q K^T : two 16x16 f32 tiles (keys kb..+15, kb+16..+31)
    v8f s0 = {}, s1 = {};
    {
      v16h b00 = *(const v16h*)&sK[buf][(0 * 16 + ln16) * 64 + 0  + half * 16];
      v16h b01 = *(const v16h*)&sK[buf][(0 * 16 + ln16) * 64 + 32 + half * 16];
      s0 = WMMA_F32_F16(qf[0], b00, s0);
      s0 = WMMA_F32_F16(qf[1], b01, s0);
      v16h b10 = *(const v16h*)&sK[buf][(1 * 16 + ln16) * 64 + 0  + half * 16];
      v16h b11 = *(const v16h*)&sK[buf][(1 * 16 + ln16) * 64 + 32 + half * 16];
      s1 = WMMA_F32_F16(qf[0], b10, s1);
      s1 = WMMA_F32_F16(qf[1], b11, s1);
    }

    // online softmax; row m = half*8+r lives across the 16 lanes of this half
    float p0[8], p1[8], alpha[8];
#pragma unroll
    for (int r = 0; r < 8; ++r) {
      float mx = fmaxf(s0[r], s1[r]);
      mx = fmaxf(mx, __shfl_xor(mx, 1, 32));
      mx = fmaxf(mx, __shfl_xor(mx, 2, 32));
      mx = fmaxf(mx, __shfl_xor(mx, 4, 32));
      mx = fmaxf(mx, __shfl_xor(mx, 8, 32));
      float mnew = fmaxf(mrow[r], mx);
      alpha[r] = __expf(mrow[r] - mnew);
      p0[r] = __expf(s0[r] - mnew);
      p1[r] = __expf(s1[r] - mnew);
      float rs = p0[r] + p1[r];
      rs += __shfl_xor(rs, 1, 32);
      rs += __shfl_xor(rs, 2, 32);
      rs += __shfl_xor(rs, 4, 32);
      rs += __shfl_xor(rs, 8, 32);
      lrow[r] = lrow[r] * alpha[r] + rs;
      mrow[r] = mnew;
    }
#pragma unroll
    for (int dg = 0; dg < 4; ++dg)
#pragma unroll
      for (int r = 0; r < 8; ++r) O[dg][r] *= alpha[r];

    // C-layout -> A-fragment layout for P via per-wave LDS scratch
#pragma unroll
    for (int r = 0; r < 8; ++r) {
      int row = half * 8 + r;
      sP[wid][row * 32 + ln16]      = (_Float16)p0[r];
      sP[wid][row * 32 + 16 + ln16] = (_Float16)p1[r];
    }
    __syncthreads();
    v16h pf;
    {
      const _Float16* base = &sP[wid][ln16 * 32];
      v8h lo = *(const v8h*)&base[half * 16];
      v8h hi = *(const v8h*)&base[8 + half * 16];
#pragma unroll
      for (int i = 0; i < 8; ++i) { pf[i] = lo[i]; pf[i + 8] = hi[i]; }
    }

    // O += P * V
#pragma unroll
    for (int dg = 0; dg < 4; ++dg) {
      v16h vf = *(const v16h*)&sVT[buf][(dg * 16 + ln16) * 32 + half * 16];
      O[dg] = WMMA_F32_F16(pf, vf, O[dg]);
    }
  }

  float invl[8];
#pragma unroll
  for (int r = 0; r < 8; ++r) invl[r] = 1.0f / lrow[r];
#pragma unroll
  for (int dg = 0; dg < 4; ++dg)
#pragma unroll
    for (int r = 0; r < 8; ++r) {
      int qq = qbase + wid * 16 + half * 8 + r;
      int d = dg * 16 + ln16;
      Yh[((size_t)b * 2048 + qq) * 1024 + h * 64 + d] = (_Float16)(O[dg][r] * invl[r]);
    }
}

// ---------------------------------------------------------------------------
// launch
// ---------------------------------------------------------------------------
extern "C" void kernel_launch(void* const* d_in, const int* in_sizes, int n_in,
                              void* d_out, int out_size, void* d_ws, size_t ws_size,
                              hipStream_t stream) {
  const float* x    = (const float*)d_in[0];  // [2,2048,1024]
  const float* Wqkv = (const float*)d_in[1];  // [1024,3072]
  const float* Wo   = (const float*)d_in[2];  // [1024,1024]
  float* out = (float*)d_out;                 // [2,2048,1024]

  char* ws = (char*)d_ws;
  _Float16* xh     = (_Float16*)(ws + 0);          //  8 MiB: x f16 [4096,1024]
  _Float16* wqkvT  = (_Float16*)(ws + 8388608);    //  6 MiB: Wqkv^T f16 [3072,1024]
  _Float16* woT    = (_Float16*)(ws + 14680064);   //  2 MiB: Wo^T f16 [1024,1024]
  _Float16* qh     = (_Float16*)(ws + 16777216);   //  8 MiB: Q [bh,2048,64]
  _Float16* kh     = (_Float16*)(ws + 25165824);   //  8 MiB: K [bh,2048,64]
  _Float16* vT     = (_Float16*)(ws + 33554432);   //  8 MiB: V^T [bh,64,2048]
  _Float16* yh     = (_Float16*)(ws + 41943040);   //  8 MiB: attn out [4096,1024]

  cvt_f32_f16<<<2048, 256, 0, stream>>>(x, xh, 2 * 2048 * 1024);

  dim3 tb(32, 8);
  cvt_transpose_f16<<<dim3(3072 / 32, 1024 / 32), tb, 0, stream>>>(Wqkv, wqkvT, 1024, 3072);
  cvt_transpose_f16<<<dim3(1024 / 32, 1024 / 32), tb, 0, stream>>>(Wo, woT, 1024, 1024);

  dim3 gQKV(3072 / 128, 4096 / 128);  // (24, 32)
  gemm_f16f32<<<gQKV, 256, 0, stream>>>(xh, wqkvT, 4096, 3072, 1024, 0,
                                        nullptr, qh, kh, vT);

  dim3 gAtt(2048 / 128, 32);  // (16, 32)
  flash_attn<<<gAtt, 256, 0, stream>>>(qh, kh, vT, yh);

  dim3 gO(1024 / 128, 4096 / 128);  // (8, 32)
  gemm_f16f32<<<gO, 256, 0, stream>>>(yh, woT, 4096, 1024, 1024, 1,
                                      out, nullptr, nullptr, nullptr);
}
